// DynamicAutoencoder_32109175505771
// MI455X (gfx1250) — matrix operand order repaired and checked by reading, not compiled, not run on hardware
//
#include <hip/hip_runtime.h>

#define DT_STEP 0.01f

typedef __attribute__((ext_vector_type(16))) __bf16 v16bf;
typedef __attribute__((ext_vector_type(2)))  __bf16 v2bf;
typedef __attribute__((ext_vector_type(8)))  float  v8f;

union AFrag { v16bf v; unsigned int u[8]; };
union CFrag { v8f v; float f[8]; };
union PkCvt { v2bf v; unsigned int u; };
union HCvt  { __bf16 h; unsigned short s; };

#if defined(__gfx1250__)
#define ASYNC_LDS 1
#else
#define ASYNC_LDS 0
#endif

__device__ __forceinline__ unsigned int pack2_bf16(float a, float b) {
#if __has_builtin(__builtin_amdgcn_cvt_pk_bf16_f32)
    PkCvt c; c.v = __builtin_amdgcn_cvt_pk_bf16_f32(a, b);   // v_cvt_pk_bf16_f32
    return c.u;
#else
    v2bf r; r.x = (__bf16)a; r.y = (__bf16)b;                // native bf16 cvt if present
    PkCvt c; c.v = r;
    return c.u;
#endif
}

__device__ __forceinline__ unsigned short f32_to_bf16(float a) {
    HCvt c; c.h = (__bf16)a;
    return c.s;
}

__device__ __forceinline__ float fast_tanh(float x) {
#if __has_builtin(__builtin_amdgcn_tanhf)
    return __builtin_amdgcn_tanhf(x);   // v_tanh_f32 (CDNA5 trans op)
#else
    return tanhf(x);
#endif
}

#if ASYNC_LDS
// Per-lane async DMA: LDS[laddr] = MEM[gaddr]; tracked by ASYNCcnt.
// LDS byte address = low 32 bits of the generic (aperture) pointer (ISA 10.2).
__device__ __forceinline__ void async_load_b32(unsigned int* lds_ptr,
                                               const unsigned int* gptr) {
    unsigned int laddr = (unsigned int)(unsigned long long)lds_ptr;
    asm volatile("global_load_async_to_lds_b32 %0, %1, off"
                 :: "v"(laddr), "v"((unsigned long long)gptr)
                 : "memory");
}
__device__ __forceinline__ void wait_asynccnt0() {
    asm volatile("s_wait_asynccnt 0" ::: "memory");
}
#endif

// Pre-pack f32 weights W[K,N] -> Wp[K/2][N] bf16 pairs (uint): {W[2k2][n], W[2k2+1][n]}.
// N = 1<<shift.  One-time ~7 MB total; makes GEMM B-staging a single b32 load (DMA-able).
__global__ __launch_bounds__(256) void prepack_weights(
    const float* __restrict__ W, unsigned int* __restrict__ Wp,
    long total, int shift)
{
    long i      = (long)blockIdx.x * blockDim.x + threadIdx.x;
    long stride = (long)gridDim.x * blockDim.x;
    const long nmask = (1L << shift) - 1;
    for (; i < total; i += stride) {
        long n  = i & nmask;
        long k2 = i >> shift;
        const float* p = W + ((k2 * 2) << shift) + n;
        Wp[i] = pack2_bf16(p[0], p[1L << shift]);
    }
}

// C[M,N] = act(A[M,K] @ Wp[K,N](bf16 pairs) + bias[N])
// Template params (compile-time -> straight-line loaders, no EXEC divergence):
//   A_BF16: A is packed bf16 row-major;  A_MODE: 0 dense rows (r*a_stride),
//   1 targets mapping ((r/127)*128 + r%127 + 1)*a_stride;  ACT: tanh;  C_BF16: bf16 out.
// Block tile 128x128, 8 waves, wave tile 32x64 (2x4 WMMA 16x16x32 bf16),
// double-buffered LDS, one barrier per k-step; bf16 operands staged via async LDS DMA.
// Requires M%128==0, N%128==0, K%32==0 (true for all shapes here).
template <int A_BF16, int A_MODE, int ACT, int C_BF16>
__global__ __launch_bounds__(256) void gemm_bias_act(
    const void* __restrict__ A, long a_stride,
    const unsigned int* __restrict__ Wp, const float* __restrict__ bias,
    void* __restrict__ C, int M, int N, int K)
{
    __shared__ unsigned int lds_a[2][128][17]; // [buf][m][k-pair], +1 pad
    __shared__ unsigned int lds_b[2][128][17]; // [buf][n][k-pair]

    const int t     = threadIdx.x;
    const int lane  = t & 31;
    const int w     = t >> 5;
    const int wm    = w & 3;        // 4 waves along M (32 rows each)
    const int wn    = w >> 2;       // 2 waves along N (64 cols each)
    const int lhalf = lane >> 4;    // 0: lanes 0-15, 1: lanes 16-31
    const int lrow  = lane & 15;
    const int bm0   = blockIdx.y * 128;
    const int bn0   = blockIdx.x * 128;

    // loop-invariant staging coordinates (hoisted out of k-loop)
    long arow[8];                   // A row base address (elements)
    int  am[8], akk[8];             // LDS A coords
    int  bn[8], bkk[8];             // LDS B coords
    #pragma unroll
    for (int i = 0; i < 8; ++i) {
        int e  = t + i * 256;       // 0..2047
        akk[i] = e & 15; am[i] = e >> 4;
        int gm = bm0 + am[i];
        if (A_MODE == 0) arow[i] = (long)gm * a_stride;
        else             arow[i] = ((long)(gm / 127) * 128 + (gm % 127) + 1) * a_stride;
        bn[i]  = e & 127; bkk[i] = e >> 7;
    }

    v8f zero = {};
    v8f acc[2][4];
    #pragma unroll
    for (int i = 0; i < 2; ++i)
        #pragma unroll
        for (int j = 0; j < 4; ++j) acc[i][j] = zero;

    // ---- prologue: stage k-tile 0 into buffer 0
    #pragma unroll
    for (int i = 0; i < 8; ++i) {
        if (A_BF16) {
#if ASYNC_LDS
            async_load_b32(&lds_a[0][am[i]][akk[i]],
                           (const unsigned int*)A + (arow[i] >> 1) + akk[i]);
#else
            lds_a[0][am[i]][akk[i]] = ((const unsigned int*)A)[(arow[i] >> 1) + akk[i]];
#endif
        } else {
            const float* p = (const float*)A + arow[i] + 2 * akk[i];
            float2 v = *(const float2*)p;
            lds_a[0][am[i]][akk[i]] = pack2_bf16(v.x, v.y);
        }
    }
    #pragma unroll
    for (int i = 0; i < 8; ++i) {
        const unsigned int* wp = Wp + (long)bkk[i] * N + bn0 + bn[i];
#if ASYNC_LDS
        async_load_b32(&lds_b[0][bn[i]][bkk[i]], wp);
#else
        lds_b[0][bn[i]][bkk[i]] = *wp;
#endif
    }
#if ASYNC_LDS
    wait_asynccnt0();
#endif
    __syncthreads();

    for (int k0 = 0; k0 < K; k0 += 32) {
        const int  cur      = (k0 >> 5) & 1;
        const int  nxt      = cur ^ 1;
        const bool has_next = (k0 + 32) < K;      // wave-uniform
        const int  kn       = k0 + 32;

        // ---- issue next k-tile loads (overlap with WMMA below)
        float2       ra_f[8];
        unsigned int ra_u[8];
        unsigned int rb_u[8];
        if (has_next) {
            #pragma unroll
            for (int i = 0; i < 8; ++i) {
                if (A_BF16) {
#if ASYNC_LDS
                    async_load_b32(&lds_a[nxt][am[i]][akk[i]],
                                   (const unsigned int*)A + ((arow[i] + kn) >> 1) + akk[i]);
#else
                    ra_u[i] = ((const unsigned int*)A)[((arow[i] + kn) >> 1) + akk[i]];
#endif
                } else {
                    const float* p = (const float*)A + arow[i] + kn + 2 * akk[i];
                    ra_f[i] = *(const float2*)p;
                }
            }
            #pragma unroll
            for (int i = 0; i < 8; ++i) {
                const unsigned int* wp =
                    Wp + (long)((kn >> 1) + bkk[i]) * N + bn0 + bn[i];
#if ASYNC_LDS
                async_load_b32(&lds_b[nxt][bn[i]][bkk[i]], wp);
#else
                rb_u[i] = *wp;
#endif
            }
        }

        // ---- fragments per documented 16-bit A (16x32) / B (32x16) layouts
        AFrag afr[2], bfr[4];
        #pragma unroll
        for (int i = 0; i < 2; ++i) {
            int row = wm * 32 + i * 16 + lrow;
            int kb  = lhalf * 4;            // lanes 16-31 start at K=8 (pair 4)
            #pragma unroll
            for (int v = 0; v < 4; ++v) afr[i].u[v] = lds_a[cur][row][kb + v];
            #pragma unroll
            for (int v = 4; v < 8; ++v) afr[i].u[v] = lds_a[cur][row][8 + kb + (v - 4)];
        }
        #pragma unroll
        for (int j = 0; j < 4; ++j) {
            int col = wn * 64 + j * 16 + lrow;
            int kb  = lhalf * 4;            // lanes 16-31 start at K=8 (pair 4)
            #pragma unroll
            for (int v = 0; v < 4; ++v) bfr[j].u[v] = lds_b[cur][col][kb + v];
            #pragma unroll
            for (int v = 4; v < 8; ++v) bfr[j].u[v] = lds_b[cur][col][8 + kb + (v - 4)];
        }

        #pragma unroll
        for (int i = 0; i < 2; ++i)
            #pragma unroll
            for (int j = 0; j < 4; ++j)
                acc[i][j] = __builtin_amdgcn_wmma_f32_16x16x32_bf16(
                    false, afr[i].v, false, bfr[j].v, (short)0, acc[i][j], false, false);

        // ---- pack + store next A tile (f32 path only; bf16 paths are pure DMA)
        if (has_next) {
            if (!A_BF16) {
                #pragma unroll
                for (int i = 0; i < 8; ++i)
                    lds_a[nxt][am[i]][akk[i]] = pack2_bf16(ra_f[i].x, ra_f[i].y);
            }
#if !ASYNC_LDS
            if (A_BF16) {
                #pragma unroll
                for (int i = 0; i < 8; ++i) lds_a[nxt][am[i]][akk[i]] = ra_u[i];
            }
            #pragma unroll
            for (int i = 0; i < 8; ++i) lds_b[nxt][bn[i]][bkk[i]] = rb_u[i];
#endif
        }
#if ASYNC_LDS
        if (has_next) wait_asynccnt0();
#endif
        __syncthreads();
        (void)ra_u; (void)ra_f; (void)rb_u;
    }

    // ---- epilogue: bias + optional hw tanh; C/D layout: vgpr r -> M=r+8*lhalf, N=lrow
    #pragma unroll
    for (int i = 0; i < 2; ++i) {
        #pragma unroll
        for (int j = 0; j < 4; ++j) {
            CFrag cf; cf.v = acc[i][j];
            int n    = bn0 + wn * 64 + j * 16 + lrow;
            float bv = bias[n];
            #pragma unroll
            for (int r = 0; r < 8; ++r) {
                int m   = bm0 + wm * 32 + i * 16 + lhalf * 8 + r;
                float x = cf.f[r] + bv;
                if (ACT) x = fast_tanh(x);
                if (C_BF16) ((unsigned short*)C)[(long)m * N + n] = f32_to_bf16(x);
                else        ((float*)C)[(long)m * N + n] = x;
            }
        }
    }
}

// One thread per (batch, eigenpair): 2x2 rotation-scaling rollout, T steps.
// z_pred rows stored as r = b*T + t (matches z_true layout); b64 stores.
__global__ __launch_bounds__(256) void koopman_rollout(
    const float* __restrict__ z0, const float* __restrict__ p,
    float2* __restrict__ zpred2, int T)
{
    int idx = blockIdx.x * blockDim.x + threadIdx.x; // B*64
    int b = idx >> 6, n = idx & 63;
    float mu = p[b * 128 + 2 * n];
    float om = p[b * 128 + 2 * n + 1];
    float e  = expf(mu * DT_STEP);
    float a  = e * cosf(om * DT_STEP);
    float s  = e * sinf(om * DT_STEP);
    float ze = z0[b * 128 + 2 * n];
    float zo = z0[b * 128 + 2 * n + 1];
    for (int t = 0; t < T; ++t) {
        float ne = a * ze - s * zo;
        float no = s * ze + a * zo;
        zpred2[(long)(b * T + t) * 64 + n] = make_float2(ne, no);
        ze = ne; zo = no;
    }
}

// Partial sum of squared error. Y-index modes (compile-time):
// 0: same flat index; 1: targets window mapping; 2: x0 window mapping.
template <int MODE>
__global__ __launch_bounds__(256) void mse_partial_kernel(
    const float* __restrict__ X, const float* __restrict__ Y,
    long n, float* __restrict__ partial)
{
    __shared__ float red[256];
    long i      = (long)blockIdx.x * blockDim.x + threadIdx.x;
    long stride = (long)gridDim.x * blockDim.x;
    float acc = 0.f;
    for (; i < n; i += stride) {
        long yi;
        if (MODE == 0) {
            yi = i;
        } else {
            long col = i & 511;     // D = 512
            long r   = i >> 9;
            if (MODE == 1) yi = (((r / 127) * 128 + (r % 127) + 1) << 9) + col;
            else           yi = (r << 16) + col;   // r * T*D = r * 65536
        }
        float d = X[i] - Y[yi];
        acc += d * d;
    }
    red[threadIdx.x] = acc;
    __syncthreads();
    for (int s = 128; s > 0; s >>= 1) {
        if (threadIdx.x < s) red[threadIdx.x] += red[threadIdx.x + s];
        __syncthreads();
    }
    if (threadIdx.x == 0) partial[blockIdx.x] = red[0];
}

__global__ __launch_bounds__(256) void finalize_kernel(
    const float* __restrict__ part, float* __restrict__ out)
{
    __shared__ float red[3][256];
    int t = threadIdx.x;
    red[0][t] = part[t];
    red[1][t] = part[256 + t];
    red[2][t] = part[512 + t];
    __syncthreads();
    for (int s = 128; s > 0; s >>= 1) {
        if (t < s) {
            red[0][t] += red[0][t + s];
            red[1][t] += red[1][t + s];
            red[2][t] += red[2][t + s];
        }
        __syncthreads();
    }
    if (t == 0) {
        float lr = red[0][0] / (512.f * 512.f);
        float ll = red[1][0] / (65024.f * 128.f);
        float lp = red[2][0] / (65024.f * 512.f);
        out[0] = lr + ll + lp;
        out[1] = lr;
        out[2] = ll;
        out[3] = lp;
    }
}

extern "C" void kernel_launch(void* const* d_in, const int* in_sizes, int n_in,
                              void* d_out, int out_size, void* d_ws, size_t ws_size,
                              hipStream_t stream)
{
    (void)in_sizes; (void)n_in; (void)out_size; (void)ws_size;
    const float* win = (const float*)d_in[0];
    const float* eW1 = (const float*)d_in[1];
    const float* eb1 = (const float*)d_in[2];
    const float* eW2 = (const float*)d_in[3];
    const float* eb2 = (const float*)d_in[4];
    const float* eW3 = (const float*)d_in[5];
    const float* eb3 = (const float*)d_in[6];
    const float* dW1 = (const float*)d_in[7];
    const float* db1 = (const float*)d_in[8];
    const float* dW2 = (const float*)d_in[9];
    const float* db2 = (const float*)d_in[10];
    const float* dW3 = (const float*)d_in[11];
    const float* db3 = (const float*)d_in[12];
    const float* kW1 = (const float*)d_in[13];
    const float* kb1 = (const float*)d_in[14];
    const float* kW2 = (const float*)d_in[15];
    const float* kb2 = (const float*)d_in[16];
    float* out = (float*)d_out;
    float* ws  = (float*)d_ws;

    // workspace layout (float-sized slots; BUF1/BUF2/KH hold packed bf16)
    float* BUF1  = ws;                       // bf16 65024*1024
    float* BUF2  = BUF1  + 65024L * 1024;    // bf16 65024*1024
    float* Z0    = BUF2  + 65024L * 1024;    // f32 512*128
    float* KH    = Z0    + 512L * 128;       // bf16 512*128
    float* P     = KH    + 512L * 128;       // f32 512*128
    float* ZPRED = P     + 512L * 128;       // f32 65024*128
    float* ZTRUE = ZPRED + 65024L * 128;     // f32 65024*128
    float* XPRED = ZTRUE + 65024L * 128;     // f32 65024*512
    float* X0REC = XPRED + 65024L * 512;     // f32 512*512
    float* PART  = X0REC + 512L * 512;       // f32 3*256
    // packed bf16-pair weights [K/2][N] (uints)
    unsigned int* eW1p = (unsigned int*)(PART + 768);
    unsigned int* eW2p = eW1p + 256L * 1024;
    unsigned int* eW3p = eW2p + 512L * 1024;
    unsigned int* dW1p = eW3p + 512L * 128;
    unsigned int* dW2p = dW1p + 64L * 1024;
    unsigned int* dW3p = dW2p + 512L * 1024;
    unsigned int* kW1p = dW3p + 512L * 512;
    unsigned int* kW2p = kW1p + 64L * 128;

    dim3 blk(256);
#define PRE(W, WP, TOT, SH) \
    prepack_weights<<<dim3(256), blk, 0, stream>>>(W, WP, TOT, SH)
    PRE(eW1, eW1p, 256L * 1024, 10);
    PRE(eW2, eW2p, 512L * 1024, 10);
    PRE(eW3, eW3p, 512L * 128,  7);
    PRE(dW1, dW1p, 64L * 1024,  10);
    PRE(dW2, dW2p, 512L * 1024, 10);
    PRE(dW3, dW3p, 512L * 512,  9);
    PRE(kW1, kW1p, 64L * 128,   7);
    PRE(kW2, kW2p, 64L * 128,   7);
#undef PRE

#define GEMM(ABF, AMODE, ACT, CBF, A, ASTR, WP, B, C, M, N, K)                     \
    gemm_bias_act<ABF, AMODE, ACT, CBF>                                            \
        <<<dim3((N) / 128, (M) / 128), blk, 0, stream>>>(A, ASTR, WP, B, C, M, N, K)

    // encoder on x0 = window[:,0,:]   (row stride T*D = 65536)
    GEMM(0, 0, 1, 1, win,  65536, eW1p, eb1, BUF1, 512, 1024, 512);
    GEMM(1, 0, 1, 1, BUF1, 1024,  eW2p, eb2, BUF2, 512, 1024, 1024);
    GEMM(1, 0, 0, 0, BUF2, 1024,  eW3p, eb3, Z0,   512, 128,  1024);
    // decoder on z0
    GEMM(0, 0, 1, 1, Z0,   128,   dW1p, db1, BUF1, 512, 1024, 128);
    GEMM(1, 0, 1, 1, BUF1, 1024,  dW2p, db2, BUF2, 512, 1024, 1024);
    GEMM(1, 0, 0, 0, BUF2, 1024,  dW3p, db3, X0REC,512, 512,  1024);
    // koopman parameter MLP (KH kept bf16)
    GEMM(0, 0, 1, 1, Z0,   128,   kW1p, kb1, KH,   512, 128,  128);
    GEMM(1, 0, 0, 0, KH,   128,   kW2p, kb2, P,    512, 128,  128);
    // eigen rollout: 512*64 threads, 127 steps
    koopman_rollout<<<dim3((512 * 64) / 256), blk, 0, stream>>>(
        Z0, P, (float2*)ZPRED, 127);
    // decoder on z_pred (big)
    GEMM(0, 0, 1, 1, ZPRED, 128,  dW1p, db1, BUF1, 65024, 1024, 128);
    GEMM(1, 0, 1, 1, BUF1,  1024, dW2p, db2, BUF2, 65024, 1024, 1024);
    GEMM(1, 0, 0, 0, BUF2,  1024, dW3p, db3, XPRED,65024, 512,  1024);
    // encoder on targets = window[:,1:,:] (big, A_MODE 1)
    GEMM(0, 1, 1, 1, win,   512,  eW1p, eb1, BUF1, 65024, 1024, 512);
    GEMM(1, 0, 1, 1, BUF1,  1024, eW2p, eb2, BUF2, 65024, 1024, 1024);
    GEMM(1, 0, 0, 0, BUF2,  1024, eW3p, eb3, ZTRUE,65024, 128,  1024);
#undef GEMM
    // losses (deterministic two-stage reductions)
    mse_partial_kernel<2><<<dim3(256), blk, 0, stream>>>(X0REC, win,  512L * 512,   PART + 0);
    mse_partial_kernel<0><<<dim3(256), blk, 0, stream>>>(ZPRED, ZTRUE, 65024L * 128, PART + 256);
    mse_partial_kernel<1><<<dim3(256), blk, 0, stream>>>(XPRED, win,  65024L * 512, PART + 512);
    finalize_kernel<<<dim3(1), blk, 0, stream>>>(PART, out);
}
